// HiNT_49684181680758
// MI455X (gfx1250) — compile-verified, matrix operand order checked by reading, not run on hardware
//
#include <hip/hip_runtime.h>
#include <stdint.h>

// ---------------------------------------------------------------------------
// HiNT similarity kernel for MI455X (gfx1250, wave32, WMMA)
//   out[0 : Q*P)        = (query[i] == para[j]) ? 1.0f : 0.0f
//   out[Q*P : 2*Q*P)    = cosine_similarity(que_embed[i], para_embed[j])
// GEMM path: f32 -> bf16(hi)+bf16(lo) split, 3x v_wmma_f32_16x16x32_bf16
// ---------------------------------------------------------------------------

typedef __attribute__((ext_vector_type(16))) __bf16 v16bf;
typedef __attribute__((ext_vector_type(8)))  float  v8f;

#define EDIM      128          // embedding dim (K)
#define TILE_M    128
#define TILE_N    128
#define K_STAGE   32           // K per LDS stage == WMMA K
#define ROW_DW    20           // 16 packed bf16-pairs + 4 pad dwords (bank spread, 16B aligned)

static __device__ __forceinline__ uint32_t f32_to_bf16_rne(float f) {
  uint32_t x = __builtin_bit_cast(uint32_t, f);
  return (x + 0x7FFFu + ((x >> 16) & 1u)) >> 16;   // round-to-nearest-even
}
static __device__ __forceinline__ float bf16_to_f32(uint32_t h) {
  return __builtin_bit_cast(float, h << 16);
}

// ---------------------------------------------------------------------------
// Kernel 1: inverse norms.  One wave32 per row; rows [0,Q) from que_embed,
// rows [Q, Q+P) from para_embed.  inv[row] = 1 / max(||x||, 1e-8)
// ---------------------------------------------------------------------------
__global__ __launch_bounds__(256)
void hint_norms_kernel(const float* __restrict__ qe, const float* __restrict__ pe,
                       float* __restrict__ inv, int Q, int P) {
  const int lane = threadIdx.x & 31;
  const int wave = threadIdx.x >> 5;
  const int row  = blockIdx.x * 8 + wave;
  if (row >= Q + P) return;
  const float* src = (row < Q) ? (qe + (size_t)row * EDIM)
                               : (pe + (size_t)(row - Q) * EDIM);
  float4 v = *(const float4*)(src + lane * 4);
  float s = v.x * v.x + v.y * v.y + v.z * v.z + v.w * v.w;
  #pragma unroll
  for (int off = 16; off > 0; off >>= 1) s += __shfl_xor(s, off, 32);
  if (lane == 0) inv[row] = 1.0f / fmaxf(sqrtf(s), 1e-8f);
}

// ---------------------------------------------------------------------------
// Kernel 2: fused equality-matrix + cosine GEMM.
// Block = 256 threads (8 waves). Block tile: 128(M=q) x 128(N=p).
// Wave tile: 64x32  ->  4x2 accumulators of 16x16 f32.
// ---------------------------------------------------------------------------
__global__ __launch_bounds__(256)
void hint_gemm_kernel(const float* __restrict__ qe, const float* __restrict__ pe,
                      const int*  __restrict__ qid, const int* __restrict__ pid,
                      const float* __restrict__ inv_q, const float* __restrict__ inv_p,
                      float* __restrict__ out_xor, float* __restrict__ out_cos,
                      int Pdim) {
  __shared__ __align__(16) uint32_t sAhi[TILE_M * ROW_DW];
  __shared__ __align__(16) uint32_t sAlo[TILE_M * ROW_DW];
  __shared__ __align__(16) uint32_t sBhi[TILE_N * ROW_DW];
  __shared__ __align__(16) uint32_t sBlo[TILE_N * ROW_DW];
  __shared__ float sInvQ[TILE_M], sInvP[TILE_N];
  __shared__ int   sQ[TILE_M],    sP[TILE_N];

  const int t    = threadIdx.x;
  const int lane = t & 31;
  const int ln   = lane & 15;        // row-in-tile / col-in-tile
  const int h    = lane >> 4;        // half-select for K pairs
  const int blockM = blockIdx.y * TILE_M;
  const int blockN = blockIdx.x * TILE_N;

  if (t < TILE_M) {
    sQ[t]    = qid[blockM + t];
    sInvQ[t] = inv_q[blockM + t];
  } else {
    const int i = t - TILE_M;
    sP[i]    = pid[blockN + i];
    sInvP[i] = inv_p[blockN + i];
  }

  const int wave  = t >> 5;
  const int mbase = (wave & 1) * 64;   // 2 waves in M
  const int nbase = (wave >> 1) * 32;  // 4 waves in N

  v8f acc[4][2];
  #pragma unroll
  for (int i = 0; i < 4; ++i)
    #pragma unroll
    for (int j = 0; j < 2; ++j)
      acc[i][j] = (v8f){0.f, 0.f, 0.f, 0.f, 0.f, 0.f, 0.f, 0.f};

  union __align__(16) Frag { uint32_t u[8]; v16bf v; };

  #pragma unroll
  for (int ks = 0; ks < EDIM / K_STAGE; ++ks) {
    const int kBase = ks * K_STAGE;

    // ---- stage: global f32 -> split bf16 hi/lo -> LDS (packed K-pairs) ----
    #pragma unroll
    for (int i = 0; i < 8; ++i) {
      const int idx = t + i * 256;        // 0..2047 over 128 rows x 16 pairs
      const int row = idx >> 4;
      const int pr  = idx & 15;
      // A panel (que)
      {
        const float2 f = *(const float2*)(qe + (size_t)(blockM + row) * EDIM + kBase + pr * 2);
        uint32_t h0 = f32_to_bf16_rne(f.x);
        uint32_t l0 = f32_to_bf16_rne(f.x - bf16_to_f32(h0));
        uint32_t h1 = f32_to_bf16_rne(f.y);
        uint32_t l1 = f32_to_bf16_rne(f.y - bf16_to_f32(h1));
        sAhi[row * ROW_DW + pr] = h0 | (h1 << 16);
        sAlo[row * ROW_DW + pr] = l0 | (l1 << 16);
      }
      // B panel (para)
      {
        const float2 f = *(const float2*)(pe + (size_t)(blockN + row) * EDIM + kBase + pr * 2);
        uint32_t h0 = f32_to_bf16_rne(f.x);
        uint32_t l0 = f32_to_bf16_rne(f.x - bf16_to_f32(h0));
        uint32_t h1 = f32_to_bf16_rne(f.y);
        uint32_t l1 = f32_to_bf16_rne(f.y - bf16_to_f32(h1));
        sBhi[row * ROW_DW + pr] = h0 | (h1 << 16);
        sBlo[row * ROW_DW + pr] = l0 | (l1 << 16);
      }
    }
    __syncthreads();

    // ---- fragments: ISA 16-bit 16x32 layout; two b128 LDS loads each ----
    Frag aHi[4], aLo[4], bHi[2], bLo[2];
    #pragma unroll
    for (int tm = 0; tm < 4; ++tm) {
      const int o = (mbase + tm * 16 + ln) * ROW_DW + h * 4;
      *(uint4*)&aHi[tm].u[0] = *(const uint4*)&sAhi[o];
      *(uint4*)&aHi[tm].u[4] = *(const uint4*)&sAhi[o + 8];
      *(uint4*)&aLo[tm].u[0] = *(const uint4*)&sAlo[o];
      *(uint4*)&aLo[tm].u[4] = *(const uint4*)&sAlo[o + 8];
    }
    #pragma unroll
    for (int tn = 0; tn < 2; ++tn) {
      const int o = (nbase + tn * 16 + ln) * ROW_DW + h * 4;
      *(uint4*)&bHi[tn].u[0] = *(const uint4*)&sBhi[o];
      *(uint4*)&bHi[tn].u[4] = *(const uint4*)&sBhi[o + 8];
      *(uint4*)&bLo[tn].u[0] = *(const uint4*)&sBlo[o];
      *(uint4*)&bLo[tn].u[4] = *(const uint4*)&sBlo[o + 8];
    }

    // ---- bf16x3 WMMA: acc += Alo*Bhi + Ahi*Blo + Ahi*Bhi ----
    #pragma unroll
    for (int tm = 0; tm < 4; ++tm) {
      #pragma unroll
      for (int tn = 0; tn < 2; ++tn) {
        v8f c = acc[tm][tn];
        c = __builtin_amdgcn_wmma_f32_16x16x32_bf16(false, aLo[tm].v, false, bHi[tn].v,
                                                    (short)0, c, false, false);
        c = __builtin_amdgcn_wmma_f32_16x16x32_bf16(false, aHi[tm].v, false, bLo[tn].v,
                                                    (short)0, c, false, false);
        c = __builtin_amdgcn_wmma_f32_16x16x32_bf16(false, aHi[tm].v, false, bHi[tn].v,
                                                    (short)0, c, false, false);
        acc[tm][tn] = c;
      }
    }
    __syncthreads();
  }

  // ---- epilogue: scale by inverse norms; emit equality matrix ----
  #pragma unroll
  for (int tm = 0; tm < 4; ++tm) {
    #pragma unroll
    for (int tn = 0; tn < 2; ++tn) {
      #pragma unroll
      for (int r = 0; r < 8; ++r) {
        const int ml = mbase + tm * 16 + r + 8 * h;   // C layout: VGPR r -> M=r (+8 for hi lanes)
        const int nl = nbase + tn * 16 + ln;          // lane -> N
        const size_t off = (size_t)(blockM + ml) * (size_t)Pdim + (size_t)(blockN + nl);
        out_cos[off] = acc[tm][tn][r] * sInvQ[ml] * sInvP[nl];
        out_xor[off] = (sQ[ml] == sP[nl]) ? 1.0f : 0.0f;
      }
    }
  }
}

// ---------------------------------------------------------------------------
// Launch
// ---------------------------------------------------------------------------
extern "C" void kernel_launch(void* const* d_in, const int* in_sizes, int n_in,
                              void* d_out, int out_size, void* d_ws, size_t ws_size,
                              hipStream_t stream) {
  const int*   query = (const int*)d_in[0];     // token ids (int)
  const int*   para  = (const int*)d_in[1];
  const float* qe    = (const float*)d_in[2];   // [Q, 128] f32
  const float* pe    = (const float*)d_in[3];   // [P, 128] f32

  const int Q = in_sizes[0];                    // 4096
  const int P = in_sizes[1];                    // 16384

  float* inv   = (float*)d_ws;                  // [Q + P] inverse norms
  float* o_xor = (float*)d_out;                 // [Q*P] equality as float
  float* o_cos = o_xor + (size_t)Q * (size_t)P; // [Q*P] cosine similarity

  // 1) inverse row norms (one wave per row, 8 waves per block)
  const int rows = Q + P;
  hint_norms_kernel<<<(rows + 7) / 8, 256, 0, stream>>>(qe, pe, inv, Q, P);

  // 2) fused WMMA cosine GEMM + equality matrix (dims are multiples of 128)
  dim3 grid(P / TILE_N, Q / TILE_M);            // (128, 32)
  hint_gemm_kernel<<<grid, 256, 0, stream>>>(qe, pe, query, para,
                                             inv, inv + Q, o_xor, o_cos, P);
}
